// GraphAttentionEmbedding_48447231098883
// MI455X (gfx1250) — matrix operand order repaired and checked by reading, not compiled, not run on hardware
//
#include <hip/hip_runtime.h>
#include <math.h>

typedef __attribute__((ext_vector_type(16))) _Float16 v16h;
typedef __attribute__((ext_vector_type(8)))  _Float16 v8h;
typedef __attribute__((ext_vector_type(8)))  float    v8f;
typedef __attribute__((ext_vector_type(4)))  float    v4f;

#define D 128      // in/out channels
#define EDIM 160   // edge_attr dim (32 time + 128 msg)

// ---------------- fragment loaders (wave32 WMMA 16x16x32 f16) ----------------
// A (16x32, MxK): lane<16 -> row M=lane, K={k0..k0+7} in v0..3, {k0+16..k0+23} in v4..7
//                 lane>=16 -> same row pattern shifted by 8 (table in ISA 7.12.2)
__device__ __forceinline__ v16h ld_frag_a(const _Float16* p) {
  v8h lo = *(const v8h*)p;          // 8 halves, 16B load
  v8h hi = *(const v8h*)(p + 16);   // +16 halves
  v16h r;
#pragma unroll
  for (int i = 0; i < 8; ++i) { r[i] = lo[i]; r[i + 8] = hi[i]; }
  return r;
}
// B (32x16, KxN) stored as B^T row-major [N][K]: same K pattern as A (k0+8h..+7, k0+16+8h..+7)
__device__ __forceinline__ v16h ld_frag_b(const _Float16* p) {
  v8h lo = *(const v8h*)p;
  v8h hi = *(const v8h*)(p + 16);
  v16h r;
#pragma unroll
  for (int i = 0; i < 8; ++i) { r[i] = lo[i]; r[i + 8] = hi[i]; }
  return r;
}

// monotone float->uint map so unsigned atomicMax == float max (order-invariant)
__device__ __forceinline__ unsigned enc_f32(float f) {
  unsigned u = __float_as_uint(f);
  return (u >> 31) ? ~u : (u | 0x80000000u);
}
__device__ __forceinline__ float dec_f32(unsigned k) {
  unsigned u = (k & 0x80000000u) ? (k & 0x7FFFFFFFu) : ~k;
  return __uint_as_float(u);
}

// ---------------- conversion / packing kernels ----------------
__global__ void k_convert_f16(_Float16* __restrict__ dst, const float* __restrict__ src,
                              size_t n) {
  size_t i = (size_t)blockIdx.x * blockDim.x + threadIdx.x;
  if (i < n) dst[i] = (_Float16)src[i];
}

// src is [I][O] f32 (x @ W layout); dst is W^T f16 [O][I]
__global__ void k_convertT(_Float16* __restrict__ dst, const float* __restrict__ src,
                           int I, int O) {
  int t = blockIdx.x * blockDim.x + threadIdx.x;
  if (t >= I * O) return;
  int i = t / O, o = t % O;
  dst[(size_t)o * I + i] = (_Float16)src[t];
}

__global__ void k_pack_bias(float* __restrict__ dst, const float* __restrict__ bq,
                            const float* __restrict__ bk, const float* __restrict__ bv,
                            const float* __restrict__ bs) {
  int t = blockIdx.x * blockDim.x + threadIdx.x;
  if (t >= 4 * D) return;
  int sel = t >> 7, c = t & (D - 1);
  dst[t] = (sel == 0) ? bq[c] : (sel == 1) ? bk[c] : (sel == 2) ? bv[c] : bs[c];
}

__global__ void k_init_seg(unsigned* __restrict__ segmax, float* __restrict__ segsum, int n) {
  int i = blockIdx.x * blockDim.x + threadIdx.x;
  if (i < n) { segmax[i] = 0u; segsum[i] = 0.0f; }
}

// ---------------- edge attr: [cos(rel_t*w+b) | msg] in f16 ----------------
__global__ void k_edge_attr(const int* __restrict__ ei, const float* __restrict__ t,
                            const float* __restrict__ lu, const float* __restrict__ msg,
                            const float* __restrict__ tw, const float* __restrict__ tb,
                            _Float16* __restrict__ eah, int E) {
  int e = blockIdx.x * 8 + (threadIdx.x >> 5);
  if (e >= E) return;
  int lane = threadIdx.x & 31;
  int s = ei[e];                       // src node
  float rel = lu[s] - t[e];
  _Float16* row = eah + (size_t)e * EDIM;
  row[lane] = (_Float16)cosf(rel * tw[lane] + tb[lane]);   // 32 time dims, 1/lane
  const float* m = msg + (size_t)e * D;
#pragma unroll
  for (int j = 0; j < 4; ++j) row[32 + j * 32 + lane] = (_Float16)m[j * 32 + lane];
}

// ---------------- node GEMM: [N,128] x [128,512] -> q|k|v|skip ----------------
__global__ __launch_bounds__(256) void k_node_gemm(
    const _Float16* __restrict__ xh, const _Float16* __restrict__ WT,  // [512][128]
    const float* __restrict__ bias,                                     // [512]
    float* __restrict__ q, float* __restrict__ k, float* __restrict__ v,
    float* __restrict__ outp) {
  int lane = threadIdx.x & 31;
  int wave = threadIdx.x >> 5;
  int m0 = blockIdx.x * 16;
  int nt = blockIdx.y * 8 + wave;          // 0..31
  int n0 = nt * 16;
  const _Float16* pa = xh + (size_t)(m0 + (lane & 15)) * D + ((lane >> 4) << 3);
  const _Float16* pb = WT + (size_t)(n0 + (lane & 15)) * D + ((lane >> 4) << 3);
  v8f c = {};
#pragma unroll
  for (int k0 = 0; k0 < D; k0 += 32) {
    v16h a = ld_frag_a(pa + k0);
    v16h b = ld_frag_b(pb + k0);
    c = __builtin_amdgcn_wmma_f32_16x16x32_f16(false, a, false, b, (short)0, c,
                                               false, false);
  }
  int n = n0 + (lane & 15);
  int sel = n >> 7, col = n & (D - 1);
  float bval = bias[n];
  float* dst = (sel == 0) ? q : (sel == 1) ? k : (sel == 2) ? v : outp;
  int mb = m0 + ((lane >> 4) << 3);
#pragma unroll
  for (int r = 0; r < 8; ++r) dst[(size_t)(mb + r) * D + col] = c[r] + bval;
}

// ---------------- edge GEMM: [E,160] x [160,128] -> e ----------------
__global__ __launch_bounds__(256) void k_edge_gemm(
    const _Float16* __restrict__ eah, const _Float16* __restrict__ WeT,  // [128][160]
    float* __restrict__ eb) {
  int lane = threadIdx.x & 31;
  int wave = threadIdx.x >> 5;     // 8 N-tiles of 16 -> 128
  int m0 = blockIdx.x * 16;
  int n0 = wave * 16;
  const _Float16* pa = eah + (size_t)(m0 + (lane & 15)) * EDIM + ((lane >> 4) << 3);
  const _Float16* pb = WeT + (size_t)(n0 + (lane & 15)) * EDIM + ((lane >> 4) << 3);
  v8f c = {};
#pragma unroll
  for (int k0 = 0; k0 < EDIM; k0 += 32) {
    v16h a = ld_frag_a(pa + k0);
    v16h b = ld_frag_b(pb + k0);
    c = __builtin_amdgcn_wmma_f32_16x16x32_f16(false, a, false, b, (short)0, c,
                                               false, false);
  }
  int n = n0 + (lane & 15);
  int mb = m0 + ((lane >> 4) << 3);
#pragma unroll
  for (int r = 0; r < 8; ++r) eb[(size_t)(mb + r) * D + n] = c[r];
}

// ---------------- attention logits + segment max ----------------
__global__ void k_logits(const int* __restrict__ ei, const float* __restrict__ q,
                         const float* __restrict__ k, const float* __restrict__ eb,
                         float* __restrict__ alpha, unsigned* __restrict__ segmax, int E) {
  int e = blockIdx.x * 8 + (threadIdx.x >> 5);
  if (e >= E) return;
  int lane = threadIdx.x & 31;
  int s = ei[e], d = ei[E + e];
  int c = lane * 4;          // channels c..c+3 ; head = c/64 = lane>>4
  int h = lane >> 4;
  v4f qv = *(const v4f*)(q + (size_t)d * D + c);
  v4f kv = *(const v4f*)(k + (size_t)s * D + c);
  v4f ev = *(const v4f*)(eb + (size_t)e * D + c);
  float p = 0.f;
#pragma unroll
  for (int j = 0; j < 4; ++j) p += qv[j] * (kv[j] + ev[j]);
#pragma unroll
  for (int off = 1; off < 16; off <<= 1) p += __shfl_xor(p, off, 32);  // per-half
  if ((lane & 15) == 0) {
    float a = p * 0.125f;    // 1/sqrt(64)
    alpha[(size_t)e * 2 + h] = a;
    atomicMax(&segmax[(size_t)d * 2 + h], enc_f32(a));
  }
}

// ---------------- exp + segment sum ----------------
__global__ void k_expsum(const int* __restrict__ ei, const float* __restrict__ alpha,
                         const unsigned* __restrict__ segmax, float* __restrict__ ex,
                         float* __restrict__ segsum, int E) {
  int i = blockIdx.x * blockDim.x + threadIdx.x;
  if (i >= 2 * E) return;
  int e = i >> 1, h = i & 1;
  int d = ei[E + e];
  float m = dec_f32(segmax[(size_t)d * 2 + h]);
  float v = expf(alpha[i] - m);
  ex[i] = v;
  __hip_atomic_fetch_add(&segsum[(size_t)d * 2 + h], v, __ATOMIC_RELAXED,
                         __HIP_MEMORY_SCOPE_AGENT);
}

// ---------------- weighted scatter-aggregate into out ----------------
__global__ void k_aggregate(const int* __restrict__ ei, const float* __restrict__ ex,
                            const float* __restrict__ segsum, const float* __restrict__ v,
                            const float* __restrict__ eb, float* __restrict__ out, int E) {
  int e = blockIdx.x * 8 + (threadIdx.x >> 5);
  if (e >= E) return;
  int lane = threadIdx.x & 31;
  int s = ei[e], d = ei[E + e];
  int c = lane * 4, h = lane >> 4;
  float coeff = ex[(size_t)e * 2 + h] / (segsum[(size_t)d * 2 + h] + 1e-16f);
  v4f vv = *(const v4f*)(v + (size_t)s * D + c);
  v4f ev = *(const v4f*)(eb + (size_t)e * D + c);
  float* o = out + (size_t)d * D + c;
#pragma unroll
  for (int j = 0; j < 4; ++j)
    __hip_atomic_fetch_add(&o[j], coeff * (vv[j] + ev[j]), __ATOMIC_RELAXED,
                           __HIP_MEMORY_SCOPE_AGENT);
}

// ---------------- launcher ----------------
static inline size_t alignup(size_t x) { return (x + 255) & ~(size_t)255; }

extern "C" void kernel_launch(void* const* d_in, const int* in_sizes, int n_in,
                              void* d_out, int out_size, void* d_ws, size_t ws_size,
                              hipStream_t stream) {
  (void)n_in; (void)out_size; (void)ws_size;
  const float* x     = (const float*)d_in[0];
  const float* lu    = (const float*)d_in[1];
  const int*   ei    = (const int*)d_in[2];     // [2][E], values < N
  const float* t     = (const float*)d_in[3];
  const float* msg   = (const float*)d_in[4];
  const float* tw    = (const float*)d_in[5];
  const float* tb    = (const float*)d_in[6];
  const float* Wq    = (const float*)d_in[7];
  const float* bq    = (const float*)d_in[8];
  const float* Wk    = (const float*)d_in[9];
  const float* bk    = (const float*)d_in[10];
  const float* Wv    = (const float*)d_in[11];
  const float* bv    = (const float*)d_in[12];
  const float* We    = (const float*)d_in[13];
  const float* Wsk   = (const float*)d_in[14];
  const float* bsk   = (const float*)d_in[15];
  float* out = (float*)d_out;

  const int N = in_sizes[1];   // 50000
  const int E = in_sizes[3];   // 500000

  char* p = (char*)d_ws;
  auto take = [&](size_t bytes) { void* r = (void*)p; p += alignup(bytes); return r; };
  _Float16* xh    = (_Float16*)take((size_t)N * D * 2);
  _Float16* WTall = (_Float16*)take((size_t)4 * D * D * 2);   // [512][128] = qT|kT|vT|skipT
  _Float16* WeT   = (_Float16*)take((size_t)D * EDIM * 2);    // [128][160]
  float*    biasA = (float*)take((size_t)4 * D * 4);
  _Float16* eah   = (_Float16*)take((size_t)E * EDIM * 2);
  float*    qb    = (float*)take((size_t)N * D * 4);
  float*    kb    = (float*)take((size_t)N * D * 4);
  float*    vb    = (float*)take((size_t)N * D * 4);
  float*    ebuf  = (float*)take((size_t)E * D * 4);
  float*    alpha = (float*)take((size_t)E * 2 * 4);
  float*    exb   = (float*)take((size_t)E * 2 * 4);
  unsigned* segmx = (unsigned*)take((size_t)N * 2 * 4);
  float*    segsm = (float*)take((size_t)N * 2 * 4);

  // conversions / packing
  {
    size_t n = (size_t)N * D;
    k_convert_f16<<<(unsigned)((n + 255) / 256), 256, 0, stream>>>(xh, x, n);
  }
  k_convertT<<<(D * D + 255) / 256, 256, 0, stream>>>(WTall + 0 * D * D, Wq, D, D);
  k_convertT<<<(D * D + 255) / 256, 256, 0, stream>>>(WTall + 1 * D * D, Wk, D, D);
  k_convertT<<<(D * D + 255) / 256, 256, 0, stream>>>(WTall + 2 * D * D, Wv, D, D);
  k_convertT<<<(D * D + 255) / 256, 256, 0, stream>>>(WTall + 3 * D * D, Wsk, D, D);
  k_convertT<<<(EDIM * D + 255) / 256, 256, 0, stream>>>(WeT, We, EDIM, D);
  k_pack_bias<<<2, 256, 0, stream>>>(biasA, bq, bk, bv, bsk);
  k_init_seg<<<(2 * N + 255) / 256, 256, 0, stream>>>(segmx, segsm, 2 * N);

  // edge attributes
  k_edge_attr<<<(E + 7) / 8, 256, 0, stream>>>(ei, t, lu, msg, tw, tb, eah, E);

  // WMMA GEMMs (N and E are multiples of 16 here)
  k_node_gemm<<<dim3(N / 16, 4), 256, 0, stream>>>(xh, WTall, biasA, qb, kb, vb, out);
  k_edge_gemm<<<E / 16, 256, 0, stream>>>(eah, WeT, ebuf);

  // attention
  k_logits<<<(E + 7) / 8, 256, 0, stream>>>(ei, qb, kb, ebuf, alpha, segmx, E);
  k_expsum<<<(2 * E + 255) / 256, 256, 0, stream>>>(ei, alpha, segmx, exb, segsm, E);
  k_aggregate<<<(E + 7) / 8, 256, 0, stream>>>(ei, exb, segsm, vb, ebuf, out, E);
}
